// RelationEncoder_16716012716121
// MI455X (gfx1250) — matrix operand order repaired and checked by reading, not compiled, not run on hardware
//
#include <hip/hip_runtime.h>

#define PED   1024
#define NN    (PED * PED)        // 1,048,576 rows
#define EMB   32
#define HDIM  64
#define NGATE 256                // 4 * HDIM
#define NROWT (NN / 16)          // 65536 16-row tiles

typedef __attribute__((ext_vector_type(16))) _Float16 v16h;
typedef __attribute__((ext_vector_type(8)))  float    v8f;

__device__ __forceinline__ float fast_sigmoid(float x) {
  // 1 / (1 + 2^(-x*log2(e)))
  float e = __builtin_amdgcn_exp2f(-1.44269504088896340736f * x);
  return __builtin_amdgcn_rcpf(1.0f + e);
}
__device__ __forceinline__ float fast_tanh(float x) {
  // 1 - 2 / (2^(2x*log2(e)) + 1)
  float t = __builtin_amdgcn_exp2f(2.88539008177792681472f * x);
  return 1.0f - 2.0f * __builtin_amdgcn_rcpf(t + 1.0f);
}

__global__ __launch_bounds__(256)
__attribute__((amdgpu_waves_per_eu(4)))
void relation_encoder_kernel(
    const float* __restrict__ corr,    // [NN, 2]
    const float* __restrict__ ht_in,   // [NN, 64]
    const float* __restrict__ ct_in,   // [NN, 64]
    const int*   __restrict__ nei,     // [NN]
    const float* __restrict__ W_emb,   // [32, 2]
    const float* __restrict__ b_emb,   // [32]
    const float* __restrict__ W_ih,    // [256, 32]
    const float* __restrict__ b_ih,    // [256]
    const float* __restrict__ W_hh,    // [256, 64]
    const float* __restrict__ b_hh,    // [256]
    float* __restrict__ out_ht,        // [NN, 64]
    float* __restrict__ out_ct)        // [NN, 64]
{
  // B operand image: 16 tiles (4 gates x 4 col-subtiles) x 3 K-chunks,
  // each stored exactly in the per-lane WMMA B-register layout:
  //   lane (n = lane&15, hi = lane>>4), 16 halfs per lane with k = 8*hi + j (j<8), 16 + 8*hi + (j-8) (j>=8).
  __shared__ _Float16 Bsw[16 * 3 * 512];    // 48 KB
  __shared__ float biasLds[NGATE];
  __shared__ float WembLds[EMB * 2];
  __shared__ float bembLds[EMB];

  const int tid = threadIdx.x;

  // ---- cooperative fill: f32 weights -> f16 swizzled B image (L2-resident) ----
  for (int idx = tid; idx < 16 * 3 * 512; idx += 256) {
    int tileKc = idx >> 9;                 // tile*3 + kc
    int rem    = idx & 511;
    int ln     = rem >> 4;                 // lane 0..31
    int j      = rem & 15;                 // half slot 0..15
    int tile   = tileKc / 3;
    int kc     = tileKc - tile * 3;
    int gate   = tile >> 2;
    int nsub   = tile & 3;
    int n      = gate * 64 + nsub * 16 + (ln & 15);
    int k      = kc * 32 + ((j & 8) << 1) + ((ln >> 4) << 3) + (j & 7);   // global K in 0..95
    float v = (k < EMB) ? W_ih[n * EMB + k] : W_hh[n * HDIM + (k - EMB)];
    Bsw[idx] = (_Float16)v;
  }
  biasLds[tid] = b_ih[tid] + b_hh[tid];
  if (tid < EMB * 2) WembLds[tid] = W_emb[tid];
  if (tid < EMB)     bembLds[tid] = b_emb[tid];
  __syncthreads();

  const int wave   = tid >> 5;
  const int lane   = tid & 31;
  const int m      = lane & 15;   // A-layout row / C-layout column
  const int hi     = lane >> 4;
  const int gw     = blockIdx.x * 8 + wave;
  const int nWaves = gridDim.x * 8;

  for (int rt = gw; rt < NROWT; rt += nWaves) {
    const int rowBase = rt << 4;
    const int rowA    = rowBase + m;

    // ---- build A fragments (16x32 f16 each), K = [emb(32) | ht(64)] ----
    // ht values for this lane are two contiguous 8-float runs per chunk:
    //   chunk1: cols [8*hi .. 8*hi+7] and [16+8*hi .. 16+8*hi+7]
    //   chunk2: same + 32.  Load them as float4s (global_load_b128).
    v16h A[3];
    {
      const float2 c2 = *(const float2*)(corr + (size_t)rowA * 2);
      const float4* htr4 = (const float4*)(ht_in + (size_t)rowA * HDIM);
      union { float4 v[4]; float f[16]; } u1, u2;
      u1.v[0] = htr4[2 * hi + 0];
      u1.v[1] = htr4[2 * hi + 1];
      u1.v[2] = htr4[4 + 2 * hi + 0];
      u1.v[3] = htr4[4 + 2 * hi + 1];
      u2.v[0] = htr4[8 + 2 * hi + 0];
      u2.v[1] = htr4[8 + 2 * hi + 1];
      u2.v[2] = htr4[12 + 2 * hi + 0];
      u2.v[3] = htr4[12 + 2 * hi + 1];
      #pragma unroll
      for (int j = 0; j < 16; ++j) {
        // A-layout: slot j holds k = (j<8 ? j+8*hi : j+8+8*hi) within chunk
        const int kl = (j < 8) ? (j + 8 * hi) : (j + 8 + 8 * hi);
        float x = fmaxf(c2.x * WembLds[2 * kl] + c2.y * WembLds[2 * kl + 1] + bembLds[kl], 0.0f);
        A[0][j] = (_Float16)x;      // embedding chunk (k 0..31)
        A[1][j] = (_Float16)u1.f[j];  // ht cols 0..31  (k 32..63)
        A[2][j] = (_Float16)u2.f[j];  // ht cols 32..63 (k 64..95)
      }
    }

    int neiRow[8];
    #pragma unroll
    for (int r = 0; r < 8; ++r) neiRow[r] = nei[rowBase + r + 8 * hi];

    // ---- per 16-column slice of the hidden dim: gates i/f/g/o then LSTM ----
    // unroll 2: two independent WMMA/elementwise streams for ILP while
    // keeping VGPR usage under the 256 direct-address window.
    #pragma unroll 2
    for (int nsub = 0; nsub < 4; ++nsub) {
      // issue the slice's global loads first so they overlap the WMMA chain
      float ctv[8], htv[8];
      #pragma unroll
      for (int r = 0; r < 8; ++r) {
        const size_t ro = (size_t)(rowBase + r + 8 * hi) * HDIM + nsub * 16 + m;
        ctv[r] = ct_in[ro];
        htv[r] = ht_in[ro];
      }

      v8f acc[4];
      #pragma unroll
      for (int gate = 0; gate < 4; ++gate) {
        v8f c = {};
        #pragma unroll
        for (int kc = 0; kc < 3; ++kc) {
          const v16h b = *(const v16h*)&Bsw[(((gate * 4 + nsub) * 3 + kc) << 9) + lane * 16];
          c = __builtin_amdgcn_wmma_f32_16x16x32_f16(
                  /*neg_a=*/false, A[kc], /*neg_b=*/false, b,
                  /*c_mod=*/(short)0, c, /*reuse_a=*/false, /*reuse_b=*/false);
        }
        acc[gate] = c;
      }

      const float bI = biasLds[  0 + nsub * 16 + m];
      const float bF = biasLds[ 64 + nsub * 16 + m];
      const float bG = biasLds[128 + nsub * 16 + m];
      const float bO = biasLds[192 + nsub * 16 + m];

      #pragma unroll
      for (int r = 0; r < 8; ++r) {
        // C/D layout: lane (n=m, hi), VGPR r -> row = r + 8*hi, col = nsub*16 + m
        const size_t ro = (size_t)(rowBase + r + 8 * hi) * HDIM + nsub * 16 + m;
        const float iv = fast_sigmoid(acc[0][r] + bI);
        const float fv = fast_sigmoid(acc[1][r] + bF);
        const float gv = fast_tanh   (acc[2][r] + bG);
        const float ov = fast_sigmoid(acc[3][r] + bO);
        const float cn = fv * ctv[r] + iv * gv;
        const float hn = ov * fast_tanh(cn);
        const bool msk = neiRow[r] > 0;
        out_ht[ro] = msk ? hn : htv[r];
        out_ct[ro] = msk ? cn : ctv[r];
      }
    }
  }
}

extern "C" void kernel_launch(void* const* d_in, const int* in_sizes, int n_in,
                              void* d_out, int out_size, void* d_ws, size_t ws_size,
                              hipStream_t stream) {
  const float* corr  = (const float*)d_in[0];
  const float* ht    = (const float*)d_in[1];
  const float* ct    = (const float*)d_in[2];
  const int*   nei   = (const int*)  d_in[3];
  const float* W_emb = (const float*)d_in[4];
  const float* b_emb = (const float*)d_in[5];
  const float* W_ih  = (const float*)d_in[6];
  const float* b_ih  = (const float*)d_in[7];
  const float* W_hh  = (const float*)d_in[8];
  const float* b_hh  = (const float*)d_in[9];
  float* out_ht = (float*)d_out;
  float* out_ct = out_ht + (size_t)NN * HDIM;

  relation_encoder_kernel<<<2048, 256, 0, stream>>>(
      corr, ht, ct, nei, W_emb, b_emb, W_ih, b_ih, W_hh, b_hh, out_ht, out_ct);
}
